// MixedModel_1005022347910
// MI455X (gfx1250) — matrix operand order repaired and checked by reading, not compiled, not run on hardware
//
#include <hip/hip_runtime.h>
#include <stdint.h>

typedef unsigned short u16;
typedef __attribute__((ext_vector_type(16))) __bf16 v16bf;
typedef __attribute__((ext_vector_type(8)))  float  v8f;

#define D_IN  128
#define D_H   256
#define D_OUT 40
#define D_OUTP 48     // padded to 3x16 col tiles
#define D_3H  768
#define BN_EPS 1e-5f

union FragU {
    uint4 q[2];
    v16bf v;
};

__device__ __forceinline__ u16 f32_to_bf16(float f) {
    unsigned int u = __float_as_uint(f);
    unsigned int r = (u + 0x7FFFu + ((u >> 16) & 1u)) >> 16;   // round-to-nearest-even
    return (u16)r;
}

__device__ __forceinline__ int fenc(float f) {
    int i = __float_as_int(f);
    return i >= 0 ? i : (int)(i ^ 0x7FFFFFFF);
}
__device__ __forceinline__ float fdec(int i) {
    return __int_as_float(i >= 0 ? i : (int)(i ^ 0x7FFFFFFF));
}

// -------------------------------------------------------------------------
// utility kernels
// -------------------------------------------------------------------------
__global__ void zero_f32(float* __restrict__ p, long n) {
    long i = (long)blockIdx.x * blockDim.x + threadIdx.x;
    long stride = (long)gridDim.x * blockDim.x;
    for (; i < n; i += stride) p[i] = 0.0f;
}

__global__ void init_node(int* __restrict__ maxint, float* __restrict__ deg,
                          float* __restrict__ denom, int n) {
    int i = blockIdx.x * blockDim.x + threadIdx.x;
    if (i >= n) return;
    maxint[i] = INT_MIN;
    deg[i]    = 0.0f;
    denom[i]  = 0.0f;
}

__global__ void cvt_bf16(const float* __restrict__ src, u16* __restrict__ dst, long n) {
    long i = (long)blockIdx.x * blockDim.x + threadIdx.x;
    if (i >= n) return;
    dst[i] = f32_to_bf16(src[i]);
}

// W[k*Nc + n]  ->  Wt[n*K + k]  (bf16, zero padded to NcPad columns)
__global__ void transpose_w(const float* __restrict__ W, u16* __restrict__ Wt,
                            int K, int Nc, int NcPad) {
    long i = (long)blockIdx.x * blockDim.x + threadIdx.x;
    if (i >= (long)NcPad * K) return;
    int n = (int)(i / K);
    int k = (int)(i % K);
    float v = (n < Nc) ? W[(long)k * Nc + n] : 0.0f;
    Wt[(long)n * K + k] = f32_to_bf16(v);
}

// -------------------------------------------------------------------------
// WMMA GEMM with LDS-staged B panel + async global->LDS copy:
//   C[M, cols] = act( A[M,KDIM] @ Bt[cols,KDIM]^T + bias )
//   A: bf16 row-major, streamed from global with register double-buffering
//   Bt: bf16 N-major; one (NT*16 x KDIM) panel async-copied to LDS per block
//   block = 256 threads = 8 waves; wave -> one 16-row M-tile
// -------------------------------------------------------------------------
template<int KDIM, int NT, bool HASBIAS, bool RELU, bool OUTF32>
__global__ __launch_bounds__(256)
void wmma_gemm(const u16* __restrict__ A, const u16* __restrict__ Bt,
               const float* __restrict__ bias,
               float* __restrict__ Cf, u16* __restrict__ Cb,
               int M, int ldc, int col_off, int nvalid) {
    __shared__ __align__(16) u16 bpan[NT * 16 * KDIM];

    // ---- cooperative async copy of the contiguous B panel into LDS ----
    const int colbase = blockIdx.y * NT * 16;
    const u16* gpan = Bt + (size_t)colbase * KDIM;
    const int nchunks = NT * 16 * KDIM / 8;            // 16-byte chunks
    for (int c = threadIdx.x; c < nchunks; c += blockDim.x) {
        unsigned int ldsa = (unsigned int)(uintptr_t)(&bpan[c * 8]);
        unsigned long long ga = (unsigned long long)(uintptr_t)(gpan + c * 8);
        asm volatile("global_load_async_to_lds_b128 %0, %1, off"
                     :: "v"(ldsa), "v"(ga) : "memory");
    }
    asm volatile("s_wait_asynccnt 0x0" ::: "memory");
    __syncthreads();

    const int wave  = threadIdx.x >> 5;
    const int lane  = threadIdx.x & 31;
    const int mtile = blockIdx.x * 8 + wave;
    if (mtile * 16 < M) {
        const int half = lane >> 4;      // 0|1 : which half-wave
        const int mr   = lane & 15;      // row within A tile
        const int nc   = lane & 15;      // col within B/C tile
        int row = mtile * 16 + mr;
        if (row >= M) row = M - 1;
        const u16* Arow = A + (size_t)row * KDIM;

        v8f acc[NT];
#pragma unroll
        for (int t = 0; t < NT; ++t)
#pragma unroll
            for (int e = 0; e < 8; ++e) acc[t][e] = 0.0f;

        // A 16x32 bf16 layout: lane(half): e0..7 -> K=k0+8*half+e ; e8..15 -> K=k0+16+8*half+e-8
        FragU a_cur, a_nxt;
        a_cur.q[0] = *(const uint4*)(Arow + 8 * half);
        a_cur.q[1] = *(const uint4*)(Arow + 16 + 8 * half);

#pragma unroll
        for (int k0 = 0; k0 < KDIM; k0 += 32) {
            if (k0 + 32 < KDIM) {        // prefetch next A fragment (double buffer)
                a_nxt.q[0] = *(const uint4*)(Arow + k0 + 32 + 8 * half);
                a_nxt.q[1] = *(const uint4*)(Arow + k0 + 48 + 8 * half);
            }
            FragU b[NT];                 // B 32x16 layout: e0..7 -> K=k0+8*half+e ; e8..15 -> K=k0+16+8*half+e-8
#pragma unroll
            for (int t = 0; t < NT; ++t) {
                const u16* bc = &bpan[(size_t)(t * 16 + nc) * KDIM + k0 + 8 * half];
                b[t].q[0] = *(const uint4*)(bc);
                b[t].q[1] = *(const uint4*)(bc + 16);
            }
#pragma unroll
            for (int t = 0; t < NT; ++t)
                acc[t] = __builtin_amdgcn_wmma_f32_16x16x32_bf16(
                    false, a_cur.v, false, b[t].v, (short)0, acc[t], false, false);
            a_cur = a_nxt;
        }

#pragma unroll
        for (int t = 0; t < NT; ++t) {
            int col = colbase + t * 16 + nc;
            float bv = 0.0f;
            if (HASBIAS) bv = (col < nvalid) ? bias[col] : 0.0f;
#pragma unroll
            for (int r = 0; r < 8; ++r) {
                int orow = mtile * 16 + r + 8 * half;   // C/D layout: M = r + 8*half
                if (orow >= M) continue;
                float v = acc[t][r] + bv;
                if (RELU) v = fmaxf(v, 0.0f);
                if (OUTF32) {
                    if (col < nvalid) Cf[(size_t)orow * ldc + col_off + col] = v;
                } else {
                    Cb[(size_t)orow * ldc + col] = f32_to_bf16(v);
                }
            }
        }
    }
}

// -------------------------------------------------------------------------
// GAT attention pieces
// -------------------------------------------------------------------------
__global__ __launch_bounds__(256)
void att_scores(const float* __restrict__ xs, const float* __restrict__ asrc,
                const float* __restrict__ adst, float* __restrict__ as_,
                float* __restrict__ ad_, int n) {
    int wave = (blockIdx.x * blockDim.x + threadIdx.x) >> 5;
    int lane = threadIdx.x & 31;
    if (wave >= n) return;
    const float* xr = xs + (long)wave * D_H;
    float s0 = 0.0f, s1 = 0.0f;
#pragma unroll
    for (int j = 0; j < 8; ++j) {
        float v = xr[lane + 32 * j];
        s0 += v * asrc[lane + 32 * j];
        s1 += v * adst[lane + 32 * j];
    }
#pragma unroll
    for (int m = 16; m; m >>= 1) {
        s0 += __shfl_xor(s0, m, 32);
        s1 += __shfl_xor(s1, m, 32);
    }
    if (lane == 0) { as_[wave] = s0; ad_[wave] = s1; }
}

__global__ void deg_count(const int* __restrict__ ei, float* __restrict__ deg, int E, int n) {
    int i = blockIdx.x * blockDim.x + threadIdx.x;
    if (i >= E + n) return;
    int d = (i < E) ? ei[E + i] : (i - E);
    atomicAdd(&deg[d], 1.0f);
}

__global__ void deg_inv_sqrt(const float* __restrict__ deg, float* __restrict__ dis, int n) {
    int i = blockIdx.x * blockDim.x + threadIdx.x;
    if (i >= n) return;
    dis[i] = rsqrtf(fmaxf(deg[i], 1.0f));
}

__global__ void edge_logits(const int* __restrict__ ei, const float* __restrict__ as_,
                            const float* __restrict__ ad_, float* __restrict__ ebuf,
                            int* __restrict__ maxint, int E, int n) {
    int i = blockIdx.x * blockDim.x + threadIdx.x;
    if (i >= E + n) return;
    int s = (i < E) ? ei[i]     : (i - E);
    int d = (i < E) ? ei[E + i] : (i - E);
    float e = as_[s] + ad_[d];
    e = (e > 0.0f) ? e : 0.2f * e;          // leaky_relu(0.2)
    ebuf[i] = e;
    atomicMax(&maxint[d], fenc(e));
}

__global__ void edge_exp(const int* __restrict__ ei, const int* __restrict__ maxint,
                         float* __restrict__ ebuf, float* __restrict__ denom, int E, int n) {
    int i = blockIdx.x * blockDim.x + threadIdx.x;
    if (i >= E + n) return;
    int d = (i < E) ? ei[E + i] : (i - E);
    float w = __expf(ebuf[i] - fdec(maxint[d]));
    ebuf[i] = w;
    atomicAdd(&denom[d], w);
}

// one wave per edge: xc[d, 0:256] += alpha * xs[s, :]
__global__ __launch_bounds__(256)
void gat_scatter(const int* __restrict__ ei, const float* __restrict__ ebuf,
                 const float* __restrict__ denom, const float* __restrict__ xs,
                 float* __restrict__ xc, int E, int n) {
    long w = ((long)blockIdx.x * blockDim.x + threadIdx.x) >> 5;
    int lane = threadIdx.x & 31;
    if (w >= (long)E + n) return;
    int s = (w < E) ? ei[w]     : (int)(w - E);
    int d = (w < E) ? ei[E + w] : (int)(w - E);
    float alpha = ebuf[w] / denom[d];
    const float* xr = xs + (long)s * D_H;
    float* o = xc + (long)d * D_3H;
#pragma unroll
    for (int j = 0; j < 8; ++j) {
        int c = lane + 32 * j;
        atomicAdd(&o[c], alpha * xr[c]);
    }
}

// one wave per edge: xc[d, 512:768] += norm * xw[s, :]
__global__ __launch_bounds__(256)
void gcn_scatter(const int* __restrict__ ei, const float* __restrict__ dis,
                 const float* __restrict__ xw, float* __restrict__ xc, int E, int n) {
    long w = ((long)blockIdx.x * blockDim.x + threadIdx.x) >> 5;
    int lane = threadIdx.x & 31;
    if (w >= (long)E + n) return;
    int s = (w < E) ? ei[w]     : (int)(w - E);
    int d = (w < E) ? ei[E + w] : (int)(w - E);
    float norm = dis[s] * dis[d];
    const float* xr = xw + (long)s * D_H;
    float* o = xc + (long)d * D_3H + 2 * D_H;
#pragma unroll
    for (int j = 0; j < 8; ++j) {
        int c = lane + 32 * j;
        atomicAdd(&o[c], norm * xr[c]);
    }
}

// one wave per edge (no self loops): agg[d, 0:128] += x[s, :]
__global__ __launch_bounds__(256)
void gin_scatter(const int* __restrict__ ei, const float* __restrict__ x,
                 float* __restrict__ agg, int E) {
    long w = ((long)blockIdx.x * blockDim.x + threadIdx.x) >> 5;
    int lane = threadIdx.x & 31;
    if (w >= E) return;
    int s = ei[w];
    int d = ei[E + w];
    const float* xr = x + (long)s * D_IN;
    float* o = agg + (long)d * D_IN;
#pragma unroll
    for (int j = 0; j < 4; ++j) {
        int c = lane + 32 * j;
        atomicAdd(&o[c], xr[c]);
    }
}

__global__ void add_cvt(const float* __restrict__ x, const float* __restrict__ agg,
                        u16* __restrict__ hbf, long n) {
    long i = (long)blockIdx.x * blockDim.x + threadIdx.x;
    if (i >= n) return;
    hbf[i] = f32_to_bf16(x[i] + agg[i]);
}

// -------------------------------------------------------------------------
// BatchNorm (training-mode batch stats)
// -------------------------------------------------------------------------
__global__ __launch_bounds__(256)
void bn_stats(const float* __restrict__ xc, float* __restrict__ sums,
              float* __restrict__ sumsq, int n) {
    int r0 = blockIdx.x * 64;
    float s[3] = {0, 0, 0}, q[3] = {0, 0, 0};
    for (int r = 0; r < 64; ++r) {
        int row = r0 + r;
        if (row >= n) break;
        const float* p = xc + (long)row * D_3H;
#pragma unroll
        for (int j = 0; j < 3; ++j) {
            float v = p[threadIdx.x + 256 * j];
            s[j] += v;
            q[j] += v * v;
        }
    }
#pragma unroll
    for (int j = 0; j < 3; ++j) {
        atomicAdd(&sums[threadIdx.x + 256 * j], s[j]);
        atomicAdd(&sumsq[threadIdx.x + 256 * j], q[j]);
    }
}

__global__ void bn_final(const float* __restrict__ sums, const float* __restrict__ sumsq,
                         const float* __restrict__ gamma, const float* __restrict__ beta,
                         float* __restrict__ scale, float* __restrict__ shift,
                         float invN, int C) {
    int c = blockIdx.x * blockDim.x + threadIdx.x;
    if (c >= C) return;
    float mu  = sums[c] * invN;
    float var = sumsq[c] * invN - mu * mu;
    float sc  = gamma[c] * rsqrtf(var + BN_EPS);
    scale[c] = sc;
    shift[c] = beta[c] - mu * sc;
}

__global__ void bn_apply(const float* __restrict__ xc, const float* __restrict__ scale,
                         const float* __restrict__ shift, u16* __restrict__ xb, long total) {
    long i = (long)blockIdx.x * blockDim.x + threadIdx.x;
    if (i >= total) return;
    int c = (int)(i % D_3H);
    float v = fmaf(xc[i], scale[c], shift[c]);
    xb[i] = f32_to_bf16(fmaxf(v, 0.0f));   // relu(BN(xc))
}

// -------------------------------------------------------------------------
// host launcher
// -------------------------------------------------------------------------
extern "C" void kernel_launch(void* const* d_in, const int* in_sizes, int n_in,
                              void* d_out, int out_size, void* d_ws, size_t ws_size,
                              hipStream_t stream) {
    const float* x        = (const float*)d_in[0];
    const int*   ei       = (const int*)  d_in[1];
    const float* W_gat    = (const float*)d_in[2];
    const float* att_src  = (const float*)d_in[3];
    const float* att_dst  = (const float*)d_in[4];
    // d_in[5] = b_gat   : cancels under training-mode BN (per-channel shift)
    const float* W_gin1   = (const float*)d_in[6];
    const float* b_gin1   = (const float*)d_in[7];
    const float* W_gin2   = (const float*)d_in[8];
    // d_in[9] = b_gin2  : cancels under BN
    const float* W_gcn    = (const float*)d_in[10];
    // d_in[11] = b_gcn  : cancels under BN
    const float* gamma    = (const float*)d_in[12];
    const float* beta     = (const float*)d_in[13];
    const float* W_lin    = (const float*)d_in[14];
    const float* b_lin    = (const float*)d_in[15];
    const float* W_lin2   = (const float*)d_in[16];
    const float* b_lin2   = (const float*)d_in[17];
    float* out = (float*)d_out;

    const int N  = in_sizes[0] / D_IN;
    const int E  = in_sizes[1] / 2;
    const int TE = E + N;

    // ---- workspace carve-out ----
    char* ws = (char*)d_ws;
    size_t off = 0;
    auto alloc = [&](size_t bytes) -> void* {
        off = (off + 255) & ~(size_t)255;
        void* p = ws + off;
        off += bytes;
        return p;
    };
    float* xs     = (float*)alloc((size_t)N * D_H * 4);     // x @ W_gat
    float* xw     = (float*)alloc((size_t)N * D_H * 4);     // x @ W_gcn
    u16*   xbf    = (u16*)  alloc((size_t)N * D_IN * 2);    // bf16(x)
    float* agg    = (float*)alloc((size_t)N * D_IN * 4);    // GIN neighbor sum
    u16*   hbf    = (u16*)  alloc((size_t)N * D_IN * 2);    // bf16(x+agg)
    u16*   hidbf  = (u16*)  alloc((size_t)N * D_H * 2);     // bf16 relu(GIN1)
    float* xc     = (float*)alloc((size_t)N * D_3H * 4);    // concat accumulator
    u16*   xbbf   = (u16*)  alloc((size_t)N * D_3H * 2);    // bf16 relu(BN(xc))
    u16*   h2bf   = (u16*)  alloc((size_t)N * D_H * 2);     // bf16 relu(head1)
    float* ebuf   = (float*)alloc((size_t)TE * 4);
    float* as_    = (float*)alloc((size_t)N * 4);
    float* ad_    = (float*)alloc((size_t)N * 4);
    float* deg    = (float*)alloc((size_t)N * 4);
    float* dis    = (float*)alloc((size_t)N * 4);
    float* denom  = (float*)alloc((size_t)N * 4);
    int*   maxint = (int*)  alloc((size_t)N * 4);
    float* sums   = (float*)alloc((size_t)2 * D_3H * 4);    // sums + sumsq adjacent
    float* sumsq  = sums + D_3H;
    float* bscale = (float*)alloc((size_t)D_3H * 4);
    float* bshift = (float*)alloc((size_t)D_3H * 4);
    u16* wgat_t  = (u16*)alloc((size_t)D_H * D_IN * 2);
    u16* wgcn_t  = (u16*)alloc((size_t)D_H * D_IN * 2);
    u16* wgin1_t = (u16*)alloc((size_t)D_H * D_IN * 2);
    u16* wgin2_t = (u16*)alloc((size_t)D_H * D_H * 2);
    u16* wlin_t  = (u16*)alloc((size_t)D_H * D_3H * 2);
    u16* wlin2_t = (u16*)alloc((size_t)D_OUTP * D_H * 2);

    auto cdiv = [](long a, long b) { return (int)((a + b - 1) / b); };

    // ---- init / precompute ----
    zero_f32<<<2048, 256, 0, stream>>>(xc,  (long)N * D_3H);
    zero_f32<<<2048, 256, 0, stream>>>(agg, (long)N * D_IN);
    zero_f32<<<8, 256, 0, stream>>>(sums, 2 * D_3H);
    init_node<<<cdiv(N, 256), 256, 0, stream>>>(maxint, deg, denom, N);
    cvt_bf16<<<cdiv((long)N * D_IN, 256), 256, 0, stream>>>(x, xbf, (long)N * D_IN);
    transpose_w<<<cdiv(D_H  * D_IN, 256), 256, 0, stream>>>(W_gat,  wgat_t,  D_IN, D_H, D_H);
    transpose_w<<<cdiv(D_H  * D_IN, 256), 256, 0, stream>>>(W_gcn,  wgcn_t,  D_IN, D_H, D_H);
    transpose_w<<<cdiv(D_H  * D_IN, 256), 256, 0, stream>>>(W_gin1, wgin1_t, D_IN, D_H, D_H);
    transpose_w<<<cdiv(D_H  * D_H,  256), 256, 0, stream>>>(W_gin2, wgin2_t, D_H,  D_H, D_H);
    transpose_w<<<cdiv(D_H  * D_3H, 256), 256, 0, stream>>>(W_lin,  wlin_t,  D_3H, D_H, D_H);
    transpose_w<<<cdiv(D_OUTP * D_H, 256), 256, 0, stream>>>(W_lin2, wlin2_t, D_H, D_OUT, D_OUTP);

    const int mtiles = cdiv(N, 16);
    const dim3 gA(cdiv(mtiles, 8), D_H / 64);   // 8 waves/block, NT=4 -> 64 cols / grid.y

    // ---- dense transforms (WMMA bf16 -> fp32 acc, B panel in LDS) ----
    wmma_gemm<D_IN, 4, false, false, true><<<gA, 256, 0, stream>>>(
        xbf, wgat_t, nullptr, xs, nullptr, N, D_H, 0, D_H);          // xs = x @ W_gat
    wmma_gemm<D_IN, 4, false, false, true><<<gA, 256, 0, stream>>>(
        xbf, wgcn_t, nullptr, xw, nullptr, N, D_H, 0, D_H);          // xw = x @ W_gcn

    // ---- GAT attention ----
    att_scores<<<cdiv(N, 8), 256, 0, stream>>>(xs, att_src, att_dst, as_, ad_, N);
    deg_count<<<cdiv(TE, 256), 256, 0, stream>>>(ei, deg, E, N);
    deg_inv_sqrt<<<cdiv(N, 256), 256, 0, stream>>>(deg, dis, N);
    edge_logits<<<cdiv(TE, 256), 256, 0, stream>>>(ei, as_, ad_, ebuf, maxint, E, N);
    edge_exp<<<cdiv(TE, 256), 256, 0, stream>>>(ei, maxint, ebuf, denom, E, N);

    // ---- scatter passes (wave per edge) ----
    gat_scatter<<<cdiv((long)TE * 32, 256), 256, 0, stream>>>(ei, ebuf, denom, xs, xc, E, N);
    gin_scatter<<<cdiv((long)E  * 32, 256), 256, 0, stream>>>(ei, x, agg, E);
    gcn_scatter<<<cdiv((long)TE * 32, 256), 256, 0, stream>>>(ei, dis, xw, xc, E, N);

    // ---- GIN MLP ----
    add_cvt<<<cdiv((long)N * D_IN, 256), 256, 0, stream>>>(x, agg, hbf, (long)N * D_IN);
    wmma_gemm<D_IN, 4, true, true, false><<<gA, 256, 0, stream>>>(
        hbf, wgin1_t, b_gin1, nullptr, hidbf, N, D_H, 0, D_H);
    wmma_gemm<D_H, 4, false, false, true><<<gA, 256, 0, stream>>>(
        hidbf, wgin2_t, nullptr, xc, nullptr, N, D_3H, D_H, D_H);    // xc[:,256:512]

    // ---- BN -> ReLU -> head ----
    bn_stats<<<cdiv(N, 64), 256, 0, stream>>>(xc, sums, sumsq, N);
    bn_final<<<cdiv(D_3H, 256), 256, 0, stream>>>(sums, sumsq, gamma, beta,
                                                  bscale, bshift, 1.0f / (float)N, D_3H);
    bn_apply<<<cdiv((long)N * D_3H, 256), 256, 0, stream>>>(xc, bscale, bshift, xbbf,
                                                            (long)N * D_3H);
    wmma_gemm<D_3H, 4, true, true, false><<<gA, 256, 0, stream>>>(
        xbbf, wlin_t, b_lin, nullptr, h2bf, N, D_H, 0, D_H);
    const dim3 gO(cdiv(mtiles, 8), 1);          // NT=3 -> 48 padded cols, 40 valid
    wmma_gemm<D_H, 3, true, false, true><<<gO, 256, 0, stream>>>(
        h2bf, wlin2_t, b_lin2, out, nullptr, N, D_OUT, 0, D_OUT);
}
